// SelfAttentionLayer_35304631173866
// MI455X (gfx1250) — compile-verified, matrix operand order checked by reading, not run on hardware
//
#include <hip/hip_runtime.h>
#include <hip/hip_bf16.h>

// ---------------------------------------------------------------------------
// Fused transformer block for MI455X (gfx1250).  softmax rows sum to 1, so the
// reference's AV term reduces to V and the O(S^2 D) attention disappears:
//   V  = X @ Wv + bv            (bf16 WMMA, f32 accum, +X residual)
//   X1 = LN(X + V)
//   H  = relu(X1 @ W1 + b1)     (bf16 WMMA, bf16 out)
//   F  = H @ W2 + b2            (bf16 WMMA, +X1 residual)
//   out= LN(X1 + F)
// GEMMs use double-buffered LDS tiles filled with global_load_async_to_lds_b128
// (ASYNCcnt) and v_wmma_f32_16x16x32_bf16 compute.
// ---------------------------------------------------------------------------

typedef __attribute__((ext_vector_type(16))) __bf16 v16bf;
typedef __attribute__((ext_vector_type(8)))  float  v8f;

static __device__ __forceinline__ unsigned short f32_to_bf16(float f) {
  unsigned u = __builtin_bit_cast(unsigned, f);
  u += 0x7FFFu + ((u >> 16) & 1u);          // round-to-nearest-even
  return (unsigned short)(u >> 16);
}
static __device__ __forceinline__ float bf16_to_f32(unsigned short h) {
  unsigned u = ((unsigned)h) << 16;
  return __builtin_bit_cast(float, u);
}

// ------------------------- conversion kernels ------------------------------

__global__ __launch_bounds__(256) void k_cvt_bf16(
    const float* __restrict__ in, unsigned short* __restrict__ out, int n) {
  int i = blockIdx.x * blockDim.x + threadIdx.x;
  if (i < n) out[i] = f32_to_bf16(in[i]);
}

// W (K x N, f32, row-major) -> Wt (N x K, bf16, row-major)
__global__ __launch_bounds__(256) void k_transpose_bf16(
    const float* __restrict__ in, unsigned short* __restrict__ out, int K, int N) {
  int i = blockIdx.x * blockDim.x + threadIdx.x;
  if (i < K * N) {
    int n = i / K;
    int k = i - n * K;
    out[(size_t)n * K + k] = f32_to_bf16(in[(size_t)k * N + n]);
  }
}

// ------------------------------ WMMA GEMM ----------------------------------
// C[m][n] = sum_k A[m][k] * Bt[n][k] + bias[n] (+residual) (+relu)
// Block tile 128x256, K-step 32, 8 waves; each wave owns a 64x64 region as a
// 4x4 grid of 16x16 v_wmma_f32_16x16x32_bf16 fragments.  Tiles are staged to
// double-buffered LDS with global_load_async_to_lds_b128 (ASYNCcnt tracked).
// LDS rows hold 16 bf16-pairs padded to 20 uints (80B, 16B-aligned stride).

union FragU { v16bf v; unsigned u[8]; };

template<int RESID, bool RELU, bool OUTBF16>   // RESID: 0 none, 1 f32, 2 bf16
__global__ __launch_bounds__(256) void k_wmma_gemm(
    const unsigned short* __restrict__ A,      // M x K  bf16
    const unsigned short* __restrict__ Bt,     // N x K  bf16 (pre-transposed)
    const float*          __restrict__ bias,   // N      f32
    const float*          __restrict__ residF, // M x N  f32   (RESID==1)
    const unsigned short* __restrict__ residH, // M x N  bf16  (RESID==2)
    float*                __restrict__ outF,   // M x N  f32   (!OUTBF16)
    unsigned short*       __restrict__ outH,   // M x N  bf16  (OUTBF16)
    int N, int K)
{
  __shared__ unsigned lA[2][128 * 20];   // 2 x 10 KB
  __shared__ unsigned lB[2][256 * 20];   // 2 x 20 KB

  const int tid     = threadIdx.x;
  const int lane    = tid & 31;
  const int wave    = tid >> 5;
  const int waveRow = wave >> 2;   // 0..1 -> 64 rows each
  const int waveCol = wave & 3;    // 0..3 -> 64 cols each
  const int rowBase = blockIdx.y * 128;
  const int colBase = blockIdx.x * 256;
  const int lr      = lane & 15;
  const int half    = lane >> 4;

  // ---- per-lane async-copy slots: 16B chunk per lane per issue ----
  unsigned long long aG[2]; unsigned aL[2];
  unsigned long long bG[4]; unsigned bL[4];
#pragma unroll
  for (int s = 0; s < 2; ++s) {            // A: 512 chunks / 8 waves = 2/lane
    int c = wave * 64 + s * 32 + lane;
    int row = c >> 2, cc = c & 3;
    aL[s] = (unsigned)(row * 80 + cc * 16);
    aG[s] = (unsigned long long)(size_t)(A + (size_t)(rowBase + row) * K + cc * 8);
  }
#pragma unroll
  for (int s = 0; s < 4; ++s) {            // B: 1024 chunks / 8 waves = 4/lane
    int c = wave * 128 + s * 32 + lane;
    int row = c >> 2, cc = c & 3;
    bL[s] = (unsigned)(row * 80 + cc * 16);
    bG[s] = (unsigned long long)(size_t)(Bt + (size_t)(colBase + row) * K + cc * 8);
  }
  const unsigned baseA[2] = {(unsigned)(size_t)&lA[0][0], (unsigned)(size_t)&lA[1][0]};
  const unsigned baseB[2] = {(unsigned)(size_t)&lB[0][0], (unsigned)(size_t)&lB[1][0]};

  auto issue_tile = [&](int buf) {
#pragma unroll
    for (int s = 0; s < 2; ++s) {
      asm volatile("global_load_async_to_lds_b128 %0, %1, off"
                   :: "v"(baseA[buf] + aL[s]), "v"(aG[s]) : "memory");
      aG[s] += 64;                         // advance one K-step (32 bf16)
    }
#pragma unroll
    for (int s = 0; s < 4; ++s) {
      asm volatile("global_load_async_to_lds_b128 %0, %1, off"
                   :: "v"(baseB[buf] + bL[s]), "v"(bG[s]) : "memory");
      bG[s] += 64;
    }
  };

  v8f acc[4][4];
#pragma unroll
  for (int i = 0; i < 4; ++i)
#pragma unroll
    for (int j = 0; j < 4; ++j)
#pragma unroll
      for (int e = 0; e < 8; ++e) acc[i][j][e] = 0.0f;

  issue_tile(0);                           // prologue prefetch

  for (int k0 = 0; k0 < K; k0 += 32) {
    const int cur = (k0 >> 5) & 1;
    asm volatile("s_wait_asynccnt 0" ::: "memory");  // my async writes landed
    __syncthreads();                                  // everyone's landed
    if (k0 + 32 < K) issue_tile(cur ^ 1);             // prefetch next K-step

    // ---- build fragments per ISA layout (16-bit A 16x32, wave32) ----
    FragU a[4], b[4];
#pragma unroll
    for (int i = 0; i < 4; ++i) {
      const unsigned* p = &lA[cur][(waveRow * 64 + i * 16 + lr) * 20 + half * 4];
      a[i].u[0] = p[0]; a[i].u[1] = p[1]; a[i].u[2] = p[2];  a[i].u[3] = p[3];
      a[i].u[4] = p[8]; a[i].u[5] = p[9]; a[i].u[6] = p[10]; a[i].u[7] = p[11];
    }
#pragma unroll
    for (int j = 0; j < 4; ++j) {
      const unsigned* p = &lB[cur][(waveCol * 64 + j * 16 + lr) * 20 + half * 4];
      b[j].u[0] = p[0]; b[j].u[1] = p[1]; b[j].u[2] = p[2];  b[j].u[3] = p[3];
      b[j].u[4] = p[8]; b[j].u[5] = p[9]; b[j].u[6] = p[10]; b[j].u[7] = p[11];
    }

#pragma unroll
    for (int i = 0; i < 4; ++i)
#pragma unroll
      for (int j = 0; j < 4; ++j)
        acc[i][j] = __builtin_amdgcn_wmma_f32_16x16x32_bf16(
            false, a[i].v, false, b[j].v, (short)0, acc[i][j], false, false);
  }

  // ---- epilogue: bias (+residual) (+relu), store f32 or bf16 ----
#pragma unroll
  for (int i = 0; i < 4; ++i) {
#pragma unroll
    for (int j = 0; j < 4; ++j) {
      int mg = rowBase + waveRow * 64 + i * 16 + half * 8;
      int ng = colBase + waveCol * 64 + j * 16 + lr;
      float bsum = bias[ng];
#pragma unroll
      for (int r = 0; r < 8; ++r) {
        size_t idx = (size_t)(mg + r) * N + ng;
        float v = acc[i][j][r] + bsum;
        if (RESID == 1) v += residF[idx];
        if (RESID == 2) v += bf16_to_f32(residH[idx]);
        if (RELU) v = v > 0.0f ? v : 0.0f;
        if (OUTBF16) outH[idx] = f32_to_bf16(v);
        else         outF[idx] = v;
      }
    }
  }
}

// ------------------------------ LayerNorm ----------------------------------
// One 256-thread block per 768-wide row.  out = (x - mu) / sqrt(mean((x-mu)^2))

template<bool OUTBF16>
__global__ __launch_bounds__(256) void k_layernorm(
    const float* __restrict__ in, float* __restrict__ outF,
    unsigned short* __restrict__ outH)
{
  constexpr int D = 768, EPT = 3;
  __shared__ float red[256];
  const int row = blockIdx.x;
  const int tid = threadIdx.x;
  const float* x = in + (size_t)row * D;

  float v[EPT];
  float s = 0.0f;
#pragma unroll
  for (int e = 0; e < EPT; ++e) { v[e] = x[tid + e * 256]; s += v[e]; }
  red[tid] = s;
  __syncthreads();
  for (int off = 128; off > 0; off >>= 1) {
    if (tid < off) red[tid] += red[tid + off];
    __syncthreads();
  }
  const float mu = red[0] * (1.0f / D);
  __syncthreads();

  float s2 = 0.0f;
#pragma unroll
  for (int e = 0; e < EPT; ++e) { float d = v[e] - mu; s2 += d * d; }
  red[tid] = s2;
  __syncthreads();
  for (int off = 128; off > 0; off >>= 1) {
    if (tid < off) red[tid] += red[tid + off];
    __syncthreads();
  }
  const float inv = 1.0f / sqrtf(red[0] * (1.0f / D));

#pragma unroll
  for (int e = 0; e < EPT; ++e) {
    float o = (v[e] - mu) * inv;
    size_t idx = (size_t)row * D + tid + e * 256;
    if (OUTBF16) outH[idx] = f32_to_bf16(o);
    else         outF[idx] = o;
  }
}

// ------------------------------ launcher -----------------------------------

extern "C" void kernel_launch(void* const* d_in, const int* in_sizes, int n_in,
                              void* d_out, int out_size, void* d_ws, size_t ws_size,
                              hipStream_t stream) {
  (void)in_sizes; (void)n_in; (void)out_size; (void)ws_size;
  const int M = 4 * 2048;   // B * S
  const int D = 768;
  const int H = 3072;

  const float* X  = (const float*)d_in[0];
  const float* Wv = (const float*)d_in[5];
  const float* bv = (const float*)d_in[6];
  const float* W1 = (const float*)d_in[7];
  const float* b1 = (const float*)d_in[8];
  const float* W2 = (const float*)d_in[9];
  const float* b2 = (const float*)d_in[10];
  float* out = (float*)d_out;

  char* ws = (char*)d_ws;
  size_t off = 0;
  auto carve = [&](size_t bytes) -> char* {
    char* p = ws + off;
    off = (off + bytes + 255) & ~(size_t)255;
    return p;
  };
  unsigned short* WvT  = (unsigned short*)carve((size_t)D * D * 2);  // D x D
  unsigned short* W1T  = (unsigned short*)carve((size_t)H * D * 2);  // H x D
  unsigned short* W2T  = (unsigned short*)carve((size_t)D * H * 2);  // D x H
  unsigned short* Xbf  = (unsigned short*)carve((size_t)M * D * 2);  // M x D
  unsigned short* X1bf = (unsigned short*)carve((size_t)M * D * 2);  // M x D
  unsigned short* H1bf = (unsigned short*)carve((size_t)M * H * 2);  // M x H
  float*          P    = (float*)carve((size_t)M * D * 4);           // M x D pre-LN

  // --- one-time precision staging (weights transposed to N x K) ---
  k_transpose_bf16<<<(D * D + 255) / 256, 256, 0, stream>>>(Wv, WvT, D, D);
  k_transpose_bf16<<<(D * H + 255) / 256, 256, 0, stream>>>(W1, W1T, D, H);
  k_transpose_bf16<<<(H * D + 255) / 256, 256, 0, stream>>>(W2, W2T, H, D);
  k_cvt_bf16<<<(M * D + 255) / 256, 256, 0, stream>>>(X, Xbf, M * D);

  // --- GEMM1: P = X @ Wv + bv + X (residual) ---
  dim3 gridV(D / 256, M / 128);
  k_wmma_gemm<1, false, false><<<gridV, 256, 0, stream>>>(
      Xbf, WvT, bv, X, nullptr, P, nullptr, D, D);

  // --- X1 = LN(P), emitted in bf16 for the next GEMMs ---
  k_layernorm<true><<<M, 256, 0, stream>>>(P, nullptr, X1bf);

  // --- GEMM2: H1 = relu(X1 @ W1 + b1), bf16 out ---
  dim3 gridF1(H / 256, M / 128);
  k_wmma_gemm<0, true, true><<<gridF1, 256, 0, stream>>>(
      X1bf, W1T, b1, nullptr, nullptr, nullptr, H1bf, H, D);

  // --- GEMM3: P = H1 @ W2 + b2 + X1 (residual) ---
  dim3 gridF2(D / 256, M / 128);
  k_wmma_gemm<2, false, false><<<gridF2, 256, 0, stream>>>(
      H1bf, W2T, b2, nullptr, X1bf, P, nullptr, D, H);

  // --- out = LN(P), f32 ---
  k_layernorm<false><<<M, 256, 0, stream>>>(P, out, nullptr);
}
